// AttentionDecoder_83966610637343
// MI455X (gfx1250) — matrix operand order repaired and checked by reading, not compiled, not run on hardware
//
#include <hip/hip_runtime.h>
#include <hip/hip_bf16.h>

typedef __attribute__((ext_vector_type(16))) __bf16 v16bf;
typedef __attribute__((ext_vector_type(8)))  float  v8f;
typedef __attribute__((ext_vector_type(4)))  unsigned int u32x4;
typedef __attribute__((ext_vector_type(8)))  unsigned int u32x8;

#define BATCH 32
#define T_ENC 512
#define HDIM  512
#define KVDIM 1024
#define OUTD  128
#define SDEC  128
#define XDIM  (OUTD + KVDIM)   // 1152
#define G4H   (4 * HDIM)       // 2048
#define CHUNK  8               // key rows per TDM transfer
#define NCHUNK (T_ENC / CHUNK) // 64

__device__ __forceinline__ float sigmoidf_(float z) { return 1.f / (1.f + __expf(-z)); }

// ---------------------------------------------------------------------------
// Tensor Data Mover: DMA CHUNK x KVDIM fp32 rows from global to LDS.
// D# per CDNA5 ISA ch.8: group0 = {count/type/lds_addr/global_addr},
// group1 = {data_size, tensor/tile dims, dim0 stride}.  Issued per-wave.
// ---------------------------------------------------------------------------
__device__ __forceinline__ void tdm_load_rows(const float* gsrc, unsigned lds_off) {
  const unsigned long long ga = (unsigned long long)(uintptr_t)gsrc;
  u32x4 g0;
  g0[0] = 1u;                                            // count=1 (valid user D#)
  g0[1] = lds_off;                                       // lds_addr [63:32]
  g0[2] = (unsigned)(ga & 0xffffffffu);                  // global_addr[31:0]
  g0[3] = (unsigned)((ga >> 32) & 0x01ffffffu) | (2u << 30); // addr[56:32] | type=2
  u32x8 g1;
  g1[0] = 0x00020000u;                 // data_size=2 (4 bytes); no multicast/pad/iterate
  g1[1] = (unsigned)KVDIM << 16;       // tensor_dim0[15:0] @ bits[63:48]
  g1[2] = (unsigned)T_ENC << 16;       // tensor_dim1[15:0] @ bits[95:80]
  g1[3] = (unsigned)KVDIM << 16;       // tile_dim0 @ bits[127:112]
  g1[4] = (unsigned)CHUNK;             // tile_dim1 @ bits[143:128]; tile_dim2=0
  g1[5] = (unsigned)KVDIM;             // tensor_dim0_stride[31:0]
  g1[6] = 0u;                          // stride[47:32]=0, tensor_dim1_stride lo=0
  g1[7] = 0u;
  asm volatile("tensor_load_to_lds %0, %1" :: "s"(g0), "s"(g1) : "memory");
}

// ---------------------------------------------------------------------------
// Accumulate one 16x16 fp32 tile of  A[16 x klen] @ W[16 x klen]^T  (bf16 WMMA).
// Lane layout: contiguous-K for both A 16x32 and B 32x16 (k = 16*(lane/16) + s).
// ---------------------------------------------------------------------------
__device__ __forceinline__ void wmma_tile_accum(const float* __restrict__ A, int lda,
                                                const float* __restrict__ W, int ldw,
                                                int klen, int lane, v8f& acc) {
  const int m  = lane & 15;
  const int g  = lane >> 4;
  const int kA = g * 16;
  const int kB = g * 16;
  for (int k0 = 0; k0 < klen; k0 += 32) {
    const float* ap = A + (size_t)m * lda + k0 + kA;
    float4 fa0 = *(const float4*)(ap);
    float4 fa1 = *(const float4*)(ap + 4);
    float4 fa2 = *(const float4*)(ap + 8);
    float4 fa3 = *(const float4*)(ap + 12);
    const float* bp = W + (size_t)m * ldw + k0 + kB;   // m==n for B lanes (l&15)
    float4 fb0 = *(const float4*)(bp);
    float4 fb1 = *(const float4*)(bp + 4);
    float4 fb2 = *(const float4*)(bp + 8);
    float4 fb3 = *(const float4*)(bp + 12);
    v16bf a, b;
    a[0]=(__bf16)fa0.x;  a[1]=(__bf16)fa0.y;  a[2]=(__bf16)fa0.z;  a[3]=(__bf16)fa0.w;
    a[4]=(__bf16)fa1.x;  a[5]=(__bf16)fa1.y;  a[6]=(__bf16)fa1.z;  a[7]=(__bf16)fa1.w;
    a[8]=(__bf16)fa2.x;  a[9]=(__bf16)fa2.y;  a[10]=(__bf16)fa2.z; a[11]=(__bf16)fa2.w;
    a[12]=(__bf16)fa3.x; a[13]=(__bf16)fa3.y; a[14]=(__bf16)fa3.z; a[15]=(__bf16)fa3.w;
    b[0]=(__bf16)fb0.x;  b[1]=(__bf16)fb0.y;  b[2]=(__bf16)fb0.z;  b[3]=(__bf16)fb0.w;
    b[4]=(__bf16)fb1.x;  b[5]=(__bf16)fb1.y;  b[6]=(__bf16)fb1.z;  b[7]=(__bf16)fb1.w;
    b[8]=(__bf16)fb2.x;  b[9]=(__bf16)fb2.y;  b[10]=(__bf16)fb2.z; b[11]=(__bf16)fb2.w;
    b[12]=(__bf16)fb3.x; b[13]=(__bf16)fb3.y; b[14]=(__bf16)fb3.z; b[15]=(__bf16)fb3.w;
    acc = __builtin_amdgcn_wmma_f32_16x16x32_bf16(false, a, false, b, (short)0, acc,
                                                  false, false);
  }
}

// C[M,N] = A1 @ W1^T (+ A2 @ W2^T) (+ b1 + b2).
// blockDim = 128 (4 waves); wave w handles N-tile = blockIdx.y*4 + w; blockIdx.x = M-tile.
__global__ void wmma_gemm_bias(const float* __restrict__ A1, int lda1, int k1,
                               const float* __restrict__ W1,
                               const float* __restrict__ A2, int lda2, int k2,
                               const float* __restrict__ W2,
                               const float* __restrict__ b1, const float* __restrict__ b2,
                               float* __restrict__ C, int ldc) {
  const int lane = threadIdx.x & 31;
  const int wave = threadIdx.x >> 5;
  const int tm = blockIdx.x;
  const int tn = blockIdx.y * 4 + wave;
  v8f acc = {};
  wmma_tile_accum(A1 + (size_t)tm * 16 * lda1, lda1,
                  W1 + (size_t)tn * 16 * k1, k1, k1, lane, acc);
  if (A2)
    wmma_tile_accum(A2 + (size_t)tm * 16 * lda2, lda2,
                    W2 + (size_t)tn * 16 * k2, k2, k2, lane, acc);
  const int n = tn * 16 + (lane & 15);
  float bias = 0.f;
  if (b1) bias += b1[n];
  if (b2) bias += b2[n];
#pragma unroll
  for (int r = 0; r < 8; ++r) {
    const int m = tm * 16 + (lane >> 4) * 8 + r;
    C[(size_t)m * ldc + n] = acc[r] + bias;
  }
}

// ---------------------------------------------------------------------------
// One workgroup per batch row. TDM double-buffers key rows into LDS while the
// VALU computes tanh-scores on the previous chunk; partial sums combined with
// ds_add_f32; then block softmax and context reduction -> x[b,128:1152].
// ---------------------------------------------------------------------------
__global__ void attention_kernel(const float* __restrict__ key,
                                 const float* __restrict__ value,
                                 const float* __restrict__ q,
                                 const float* __restrict__ We,
                                 float* __restrict__ x) {
  __shared__ float skey[2][CHUNK * KVDIM];   // 2 x 32 KB staging
  __shared__ float sq[KVDIM];
  __shared__ float swe[KVDIM];
  __shared__ float ssc[T_ENC];
  __shared__ float red[256];
  const int b = blockIdx.x;
  const int tid = threadIdx.x;   // 256 threads, 8 waves
  for (int i = tid; i < KVDIM; i += 256) { sq[i] = q[(size_t)b * KVDIM + i]; swe[i] = We[i]; }
  for (int t = tid; t < T_ENC; t += 256) ssc[t] = 0.f;

  const float* kb = key + (size_t)b * T_ENC * KVDIM;
  if (tid < 32)   // wave 0 drives the TDM
    tdm_load_rows(kb, (unsigned)(uintptr_t)&skey[0][0]);

  const int rsub = tid >> 5;          // row within chunk (0..7)
  const int ksub = (tid & 31) * 32;   // k-slice of 32 (0..992)
  for (int c = 0; c < NCHUNK; ++c) {
    if (tid < 32) {
      if (c + 1 < NCHUNK) {
        tdm_load_rows(kb + (size_t)(c + 1) * CHUNK * KVDIM,
                      (unsigned)(uintptr_t)&skey[(c + 1) & 1][0]);
        __builtin_amdgcn_s_wait_tensorcnt(1);   // chunk c landed (in-order)
      } else {
        __builtin_amdgcn_s_wait_tensorcnt(0);
      }
    }
    __syncthreads();
    const float* kr = &skey[c & 1][rsub * KVDIM + ksub];
    float s = 0.f;
#pragma unroll 4
    for (int k = 0; k < 32; ++k)
      s += tanhf(sq[ksub + k] + kr[k]) * swe[ksub + k];
    atomicAdd(&ssc[c * CHUNK + rsub], s);       // ds_add_f32
    __syncthreads();   // chunk buffer free for the next-but-one transfer
  }

  // block max
  float mloc = -1e30f;
  for (int t = tid; t < T_ENC; t += 256) mloc = fmaxf(mloc, ssc[t]);
  red[tid] = mloc; __syncthreads();
  for (int s = 128; s > 0; s >>= 1) { if (tid < s) red[tid] = fmaxf(red[tid], red[tid + s]); __syncthreads(); }
  const float mx = red[0];
  __syncthreads();
  // exp + block sum
  float sloc = 0.f;
  for (int t = tid; t < T_ENC; t += 256) { float e = __expf(ssc[t] - mx); ssc[t] = e; sloc += e; }
  red[tid] = sloc; __syncthreads();
  for (int s = 128; s > 0; s >>= 1) { if (tid < s) red[tid] += red[tid + s]; __syncthreads(); }
  const float inv = 1.f / red[0];
  __syncthreads();

  // context: each thread owns 4 consecutive d-columns of value
  float4 acc = {0.f, 0.f, 0.f, 0.f};
  const float* vb = value + (size_t)b * T_ENC * KVDIM;
  for (int t = 0; t < T_ENC; ++t) {
    if (t + 8 < T_ENC)
      __builtin_prefetch(vb + (size_t)(t + 8) * KVDIM + tid * 4, 0, 1);
    const float a = ssc[t] * inv;
    float4 v = *(const float4*)(vb + (size_t)t * KVDIM + tid * 4);
    acc.x += a * v.x; acc.y += a * v.y; acc.z += a * v.z; acc.w += a * v.w;
  }
  *(float4*)(x + (size_t)b * XDIM + OUTD + tid * 4) = acc;
}

// torch gate order i,f,g,o in chunks of H along the 4H axis; in-place c/h update.
__global__ void lstm_pointwise(const float* __restrict__ gates,
                               float* __restrict__ c, float* __restrict__ h) {
  const int idx = blockIdx.x * 256 + threadIdx.x;   // BATCH*HDIM = 16384
  const int b = idx >> 9;
  const int j = idx & (HDIM - 1);
  const float* g = gates + (size_t)b * G4H;
  const float i  = sigmoidf_(g[j]);
  const float f  = sigmoidf_(g[HDIM + j]);
  const float gg = tanhf(g[2 * HDIM + j]);
  const float o  = sigmoidf_(g[3 * HDIM + j]);
  const float cn = f * c[idx] + i * gg;
  c[idx] = cn;
  h[idx] = o * tanhf(cn);
}

// y_hat = relu(h1 @ fc_w^T + fc_b); store d_out[b,step,:] and feedback x[b,0:128]=1+y.
__global__ void fc_relu_kernel(const float* __restrict__ h1, const float* __restrict__ fc_w,
                               const float* __restrict__ fc_b, float* __restrict__ y_out,
                               float* __restrict__ x, int step) {
  const int lane = threadIdx.x & 31;
  const int wave = threadIdx.x >> 5;
  const int tm = blockIdx.x;              // 0..1  (M = 32)
  const int tn = blockIdx.y * 4 + wave;   // 0..7  (N = 128)
  v8f acc = {};
  wmma_tile_accum(h1 + (size_t)tm * 16 * HDIM, HDIM,
                  fc_w + (size_t)tn * 16 * HDIM, HDIM, HDIM, lane, acc);
  const int n = tn * 16 + (lane & 15);
  const float bias = fc_b[n];
#pragma unroll
  for (int r = 0; r < 8; ++r) {
    const int m = tm * 16 + (lane >> 4) * 8 + r;   // batch index
    float y = acc[r] + bias;
    y = y > 0.f ? y : 0.f;
    y_out[((size_t)m * SDEC + step) * OUTD + n] = y;
    x[(size_t)m * XDIM + n] = 1.f + y;
  }
}

__global__ void init_kernel(const float* __restrict__ h0, const float* __restrict__ c0,
                            float* __restrict__ hbuf, float* __restrict__ cbuf,
                            float* __restrict__ x) {
  const int idx = blockIdx.x * 256 + threadIdx.x;
  if (idx < 2 * BATCH * HDIM) { hbuf[idx] = h0[idx]; cbuf[idx] = c0[idx]; }
  if (idx < BATCH * OUTD) {
    const int b = idx / OUTD, j = idx % OUTD;
    x[(size_t)b * XDIM + j] = 0.f;    // SOS token y = 0
  }
}

extern "C" void kernel_launch(void* const* d_in, const int* in_sizes, int n_in,
                              void* d_out, int out_size, void* d_ws, size_t ws_size,
                              hipStream_t stream) {
  (void)in_sizes; (void)n_in; (void)out_size; (void)ws_size;
  const float* enc  = (const float*)d_in[0];
  const float* h0   = (const float*)d_in[1];
  const float* c0   = (const float*)d_in[2];
  const float* Wk   = (const float*)d_in[3];
  const float* Wq   = (const float*)d_in[4];
  const float* Wv   = (const float*)d_in[5];
  const float* We   = (const float*)d_in[6];
  const float* wih0 = (const float*)d_in[7];
  const float* whh0 = (const float*)d_in[8];
  const float* bih0 = (const float*)d_in[9];
  const float* bhh0 = (const float*)d_in[10];
  const float* wih1 = (const float*)d_in[11];
  const float* whh1 = (const float*)d_in[12];
  const float* bih1 = (const float*)d_in[13];
  const float* bhh1 = (const float*)d_in[14];
  const float* fc_w = (const float*)d_in[15];
  const float* fc_b = (const float*)d_in[16];
  float* out = (float*)d_out;

  // workspace carve-up
  char* ws = (char*)d_ws;
  float* key   = (float*)ws; ws += (size_t)BATCH * T_ENC * KVDIM * sizeof(float);
  float* value = (float*)ws; ws += (size_t)BATCH * T_ENC * KVDIM * sizeof(float);
  float* q     = (float*)ws; ws += (size_t)BATCH * KVDIM * sizeof(float);
  float* xbuf  = (float*)ws; ws += (size_t)BATCH * XDIM * sizeof(float);
  float* gates = (float*)ws; ws += (size_t)BATCH * G4H * sizeof(float);
  float* hbuf  = (float*)ws; ws += (size_t)2 * BATCH * HDIM * sizeof(float);
  float* cbuf  = (float*)ws; ws += (size_t)2 * BATCH * HDIM * sizeof(float);
  float* h0cur = hbuf, *h1cur = hbuf + BATCH * HDIM;
  float* c0cur = cbuf, *c1cur = cbuf + BATCH * HDIM;

  init_kernel<<<128, 256, 0, stream>>>(h0, c0, hbuf, cbuf, xbuf);

  // key/value projections: [16384 x 1024] @ [1024 x 1024]^T, bf16 WMMA
  dim3 gkv((BATCH * T_ENC) / 16, (KVDIM / 16) / 4);   // (1024, 16)
  wmma_gemm_bias<<<gkv, 128, 0, stream>>>(enc, KVDIM, KVDIM, Wk,
                                          nullptr, 0, 32, nullptr, nullptr, nullptr,
                                          key, KVDIM);
  wmma_gemm_bias<<<gkv, 128, 0, stream>>>(enc, KVDIM, KVDIM, Wv,
                                          nullptr, 0, 32, nullptr, nullptr, nullptr,
                                          value, KVDIM);

  for (int s = 0; s < SDEC; ++s) {
    // q = h[0] @ Wq^T : [32 x 512] @ [1024 x 512]^T
    wmma_gemm_bias<<<dim3(2, 16), 128, 0, stream>>>(h0cur, HDIM, HDIM, Wq,
                                                    nullptr, 0, 32, nullptr,
                                                    nullptr, nullptr, q, KVDIM);
    attention_kernel<<<BATCH, 256, 0, stream>>>(key, value, q, We, xbuf);
    // layer-0 gates = x @ wih0^T + h0 @ whh0^T + bih0 + bhh0
    wmma_gemm_bias<<<dim3(2, 32), 128, 0, stream>>>(xbuf, XDIM, XDIM, wih0,
                                                    h0cur, HDIM, HDIM, whh0,
                                                    bih0, bhh0, gates, G4H);
    lstm_pointwise<<<64, 256, 0, stream>>>(gates, c0cur, h0cur);
    // layer-1 gates = h0_new @ wih1^T + h1 @ whh1^T + bih1 + bhh1
    wmma_gemm_bias<<<dim3(2, 32), 128, 0, stream>>>(h0cur, HDIM, HDIM, wih1,
                                                    h1cur, HDIM, HDIM, whh1,
                                                    bih1, bhh1, gates, G4H);
    lstm_pointwise<<<64, 256, 0, stream>>>(gates, c1cur, h1cur);
    fc_relu_kernel<<<dim3(2, 2), 128, 0, stream>>>(h1cur, fc_w, fc_b, out, xbuf, s);
  }
}
